// PointNetEncoder_7816840479272
// MI455X (gfx1250) — compile-verified, matrix operand order checked by reading, not run on hardware
//
#include <hip/hip_runtime.h>

typedef _Float16 v16h __attribute__((ext_vector_type(16)));
typedef float    v8f  __attribute__((ext_vector_type(8)));
typedef unsigned int u32x4 __attribute__((ext_vector_type(4)));
typedef int          i32x4 __attribute__((ext_vector_type(4)));
typedef int          i32x8 __attribute__((ext_vector_type(8)));

#define GRIDN 256
#define PTS   (GRIDN*GRIDN)     // 65536
#define NLAY  18
#define NMASK 38                // B(2) * C(19)
#define NBLK  128
#define CHUNK (PTS/NBLK)        // 512 points per workgroup
#define ACC_ELEMS (NMASK*NLAY*64) // 43776

// ---- dynamic LDS layout (bytes) ----
#define OFF_WH    0
#define SZ_WH     (8*64*64*2)          // 65536: conv weights as f16
#define OFF_BIAS  (OFF_WH+SZ_WH)
#define SZ_BIAS   (9*64*4)             // 2304
#define OFF_PA    (OFF_BIAS+SZ_BIAS)
#define SZ_PA     (9*64*4)             // 2304
#define OFF_W0    (OFF_PA+SZ_PA)
#define SZ_W0     (64*2*4)             // 512
#define OFF_ACC   (OFF_W0+SZ_W0)
#define SZ_ACC    (ACC_ELEMS*4)        // 175104: masked-max accumulators
#define OFF_ZBUF  (OFF_ACC+SZ_ACC)
#define SZ_ZBUF   (8*16*64*4)          // 32768: per-wave activation tile / TDM staging
#define OFF_PM    (OFF_ZBUF+SZ_ZBUF)
#define SZ_PM     (8*16*8)             // 1024: per-wave per-point mask bitmaps
#define LDS_BYTES (OFF_PM+SZ_PM)       // 279552 < 320KB

#if defined(__HIP_DEVICE_COMPILE__) && __has_builtin(__builtin_amdgcn_tensor_load_to_lds)
#define HAVE_TDM 1
#else
#define HAVE_TDM 0
#endif

// masked-max update for one snapshot: wave's 16 points x 64 dims (lane owns
// dims lane and lane+32). Mask bits are scalarized -> SALU branches; merges
// into the shared accumulator with LDS float-max atomics (ds_max_num_f32).
__device__ inline void snap_update(float* acc, const float* zw,
                                   const unsigned long long* pmw,
                                   int l, int lane)
{
  float aLo[NMASK], aHi[NMASK];
#pragma unroll
  for (int m = 0; m < NMASK; ++m) { aLo[m] = -__builtin_inff(); aHi[m] = -__builtin_inff(); }
#pragma unroll 1
  for (int pt = 0; pt < 16; ++pt) {
    unsigned long long pm = pmw[pt];                 // uniform broadcast load
    unsigned int lo = (unsigned int)__builtin_amdgcn_readfirstlane((int)(unsigned int)pm);
    unsigned int hi = (unsigned int)__builtin_amdgcn_readfirstlane((int)(unsigned int)(pm >> 32));
    if ((lo | hi) == 0u) continue;
    const float zlo = zw[pt*64 + lane];
    const float zhi = zw[pt*64 + 32 + lane];
#pragma unroll
    for (int m = 0; m < 32; ++m)
      if ((lo >> m) & 1u) { aLo[m] = fmaxf(aLo[m], zlo); aHi[m] = fmaxf(aHi[m], zhi); }
#pragma unroll
    for (int m = 32; m < NMASK; ++m)
      if ((hi >> (m-32)) & 1u) { aLo[m] = fmaxf(aLo[m], zlo); aHi[m] = fmaxf(aHi[m], zhi); }
  }
#pragma unroll
  for (int m = 0; m < NMASK; ++m) {
    float* p0 = &acc[(m*NLAY + l)*64 + lane];
    __hip_atomic_fetch_max(p0,      aLo[m], __ATOMIC_RELAXED, __HIP_MEMORY_SCOPE_WORKGROUP);
    __hip_atomic_fetch_max(p0 + 32, aHi[m], __ATOMIC_RELAXED, __HIP_MEMORY_SCOPE_WORKGROUP);
  }
}

__global__ void __launch_bounds__(256, 1)
mlp_pool_kernel(const int* __restrict__ x,
                const float* __restrict__ conv_w0,
                const float* __restrict__ conv_w,
                const float* __restrict__ conv_b,
                const float* __restrict__ prelu_a,
                float* __restrict__ partials)
{
  extern __shared__ char smem[];
  _Float16* Wh  = (_Float16*)(smem + OFF_WH);
  float* bias   = (float*)(smem + OFF_BIAS);
  float* pa     = (float*)(smem + OFF_PA);
  float* w0s    = (float*)(smem + OFF_W0);
  float* acc    = (float*)(smem + OFF_ACC);
  float* zbuf   = (float*)(smem + OFF_ZBUF);
  unsigned long long* pmbuf = (unsigned long long*)(smem + OFF_PM);

  const int tid  = threadIdx.x;
  const int lane = tid & 31;
  const int wave = tid >> 5;

  // ---- stage conv weights into LDS as f16 ----
#if HAVE_TDM
  // Tensor Data Mover: DMA 4 x 32KB f32 chunks into the zbuf staging region,
  // then convert f32 -> f16 into Wh. One wave issues; TENSORcnt waits.
  {
    const unsigned long long gbase = (unsigned long long)(uintptr_t)conv_w;
    const unsigned int ldsoff = (unsigned int)(uintptr_t)(smem + OFF_ZBUF);
#pragma unroll 1
    for (int chunk = 0; chunk < 4; ++chunk) {
      if (tid == 0) {
        const unsigned long long ga = gbase + (unsigned long long)chunk * 32768ull;
        u32x4 g0; i32x8 g1;
        i32x4 g2 = {0,0,0,0}; i32x4 g3 = {0,0,0,0};
        i32x8 g4 = {0,0,0,0,0,0,0,0};
        // D# group0: count=1, lds_addr, 57b global addr, type=2
        g0[0] = 1u;
        g0[1] = ldsoff;
        g0[2] = (unsigned int)ga;
        g0[3] = (unsigned int)((ga >> 32) & 0x01FFFFFFull) | (2u << 30);
        // D# group1: data_size=4B; tensor_dim0=8192; tensor_dim1=1;
        // tile_dim0=8192; tile_dim1=1; tensor_dim0_stride=8192
        g1[0] = (int)(2u << 16);
        g1[1] = (int)(8192u << 16);                 // tensor_dim0[15:0]
        g1[2] = (int)((8192u >> 16) | (1u << 16));  // tensor_dim0[31:16] | tensor_dim1[15:0]
        g1[3] = (int)(8192u << 16);                 // tile_dim0
        g1[4] = 1;                                  // tile_dim1
        g1[5] = 8192;                               // tensor_dim0_stride[31:0]
        g1[6] = 0; g1[7] = 0;
        __builtin_amdgcn_tensor_load_to_lds(g0, g1, g2, g3, g4, 0);
        __builtin_amdgcn_s_wait_tensorcnt(0);
      }
      __syncthreads();
      const float* stage = (const float*)(smem + OFF_ZBUF);
      for (int i = tid; i < 8192; i += 256)
        Wh[chunk*8192 + i] = (_Float16)stage[i];
      __syncthreads();
    }
  }
#else
  for (int i = tid; i < 8*64*64; i += 256) Wh[i] = (_Float16)conv_w[i];
#endif
  for (int i = tid; i < 9*64;    i += 256) { bias[i] = conv_b[i]; pa[i] = prelu_a[i]; }
  for (int i = tid; i < 128;     i += 256) w0s[i] = conv_w0[i];
  for (int i = tid; i < ACC_ELEMS; i += 256) acc[i] = -__builtin_inff();
  __syncthreads();

  float* zw = zbuf + wave*(16*64);
  unsigned long long* pmw = pmbuf + wave*16;

#pragma unroll 1
  for (int pass = 0; pass < CHUNK/128; ++pass) {
    const int pbase = blockIdx.x*CHUNK + pass*128 + wave*16;

    // per-point 38-bit mask bitmap for this wave's 16 points
    if (lane < 16) {
      const int p = pbase + lane;
      unsigned long long pm = 0ull;
#pragma unroll 1
      for (int m = 0; m < NMASK; ++m)
        pm |= (unsigned long long)(x[m*PTS + p] == 1) << m;
      pmw[lane] = pm;
    }
    // prefetch next pass's mask words toward L2 (global_prefetch_b8)
    if (lane < 16 && pass + 1 < CHUNK/128) {
      const int pn = pbase + 128 + lane;
#pragma unroll 1
      for (int m = 0; m < NMASK; ++m)
        __builtin_prefetch(&x[m*PTS + pn], 0, 1);
    }

    // layer 0: (gx,gy) -> 64, pre-activation into zw
    for (int idx = lane; idx < 16*64; idx += 32) {
      const int pt = idx >> 6, d = idx & 63;
      const int p  = pbase + pt;
      const float px = -1.0f + (2.0f/255.0f)*(float)(p & 255);
      const float py = -1.0f + (2.0f/255.0f)*(float)((p >> 8) & 255);
      zw[idx] = px*w0s[d*2+0] + py*w0s[d*2+1] + bias[d];
    }
    snap_update(acc, zw, pmw, 0, lane);
    for (int idx = lane; idx < 16*64; idx += 32) {
      const int d = idx & 63;
      const float v = zw[idx];
      zw[idx] = v > 0.0f ? v : pa[d]*v;
    }
    snap_update(acc, zw, pmw, 1, lane);

#pragma unroll 1
    for (int L = 0; L < 8; ++L) {
      const _Float16* Wl = Wh + L*4096;
      const int row = lane & 15;
      const int hi8 = (lane >> 4) * 8;

      // A fragments: 16x32 f16, two K-chunks (K=0..31, 32..63)
      v16h a0, a1;
#pragma unroll
      for (int e = 0; e < 8; ++e) {
        a0[e]   = (_Float16)zw[row*64 +      hi8 + e];
        a0[e+8] = (_Float16)zw[row*64 + 16 + hi8 + e];
        a1[e]   = (_Float16)zw[row*64 + 32 + hi8 + e];
        a1[e+8] = (_Float16)zw[row*64 + 48 + hi8 + e];
      }

      v8f c[4] = {};
#pragma unroll
      for (int t = 0; t < 4; ++t) {
        const int j = (lane & 15) + 16*t;   // output dim owned by this lane
        v16h b0, b1;
#pragma unroll
        for (int e = 0; e < 8; ++e) {
          b0[e]   = Wl[j*64 +      hi8 + e];
          b0[e+8] = Wl[j*64 + 16 + hi8 + e];
          b1[e]   = Wl[j*64 + 32 + hi8 + e];
          b1[e+8] = Wl[j*64 + 48 + hi8 + e];
        }
        c[t] = __builtin_amdgcn_wmma_f32_16x16x32_f16(false, a0, false, b0, (short)0, c[t], false, false);
        c[t] = __builtin_amdgcn_wmma_f32_16x16x32_f16(false, a1, false, b1, (short)0, c[t], false, false);
      }

      // D-layout -> zw (pre-activation with bias)
      const int hi = lane >> 4;
      const int nl_ = lane & 15;
#pragma unroll
      for (int t = 0; t < 4; ++t) {
        const float bN = bias[(L+1)*64 + nl_ + 16*t];
#pragma unroll
        for (int r = 0; r < 8; ++r)
          zw[(r + 8*hi)*64 + (nl_ + 16*t)] = c[t][r] + bN;
      }
      snap_update(acc, zw, pmw, 2*L+2, lane);
      for (int idx = lane; idx < 16*64; idx += 32) {
        const int d = idx & 63;
        const float v = zw[idx];
        zw[idx] = v > 0.0f ? v : pa[(L+1)*64 + d]*v;
      }
      snap_update(acc, zw, pmw, 2*L+3, lane);
    }
  }

  __syncthreads();
  float* outp = partials + (size_t)blockIdx.x * ACC_ELEMS;
  for (int i = tid; i < ACC_ELEMS; i += 256) outp[i] = acc[i];
}

// snapshots of the zero point (0,0) for the empty-mask fallback
__global__ void zsnap_kernel(const float* __restrict__ conv_w,
                             const float* __restrict__ conv_b,
                             const float* __restrict__ prelu_a,
                             float* __restrict__ zsnap)
{
  __shared__ float z[64];
  const int d = threadIdx.x;   // 64 threads
  float v = conv_b[d];         // (0,0) @ W0^T + b0 = b0
  zsnap[d] = v;
  v = v > 0.0f ? v : prelu_a[d]*v;
  zsnap[64 + d] = v;
  z[d] = v;
  __syncthreads();
  for (int i = 0; i < 8; ++i) {
    float s = conv_b[(i+1)*64 + d];
    for (int k = 0; k < 64; ++k) s += z[k]*conv_w[(i*64 + d)*64 + k];
    __syncthreads();
    zsnap[(2*i+2)*64 + d] = s;
    const float pv = s > 0.0f ? s : prelu_a[(i+1)*64 + d]*s;
    zsnap[(2*i+3)*64 + d] = pv;
    z[d] = pv;
    __syncthreads();
  }
}

__global__ void reduce_pool_kernel(const float* __restrict__ partials,
                                   const float* __restrict__ zsnap,
                                   float* __restrict__ pooled)
{
  const int idx = blockIdx.x*256 + threadIdx.x;
  if (idx >= ACC_ELEMS) return;
  float m = -__builtin_inff();
  for (int b = 0; b < NBLK; ++b)
    m = fmaxf(m, partials[(size_t)b*ACC_ELEMS + idx]);
  const int d  = idx & 63;
  const int l  = (idx >> 6) % NLAY;
  const int mk = idx / (NLAY*64);
  if (m == -__builtin_inff()) m = zsnap[l*64 + d];   // empty mask -> zero-point snap
  const int bb = mk / 19, c = mk % 19;
  pooled[(((bb*NLAY) + l)*19 + c)*64 + d] = m;
}

__global__ void __launch_bounds__(256)
fc_kernel(const float* __restrict__ pooled,
          const float* __restrict__ fc_w,
          const float* __restrict__ fc_b,
          float* __restrict__ out)
{
  __shared__ float f0[1216], f1[1216];
  const int l   = blockIdx.x;   // output layer index
  const int lp  = 17 - l;       // source layer (reversal cancels in pairing)
  const int tid = threadIdx.x;
  for (int i = tid; i < 1216; i += 256) {
    f0[i] = pooled[(0*NLAY + lp)*1216 + i];
    f1[i] = pooled[(1*NLAY + lp)*1216 + i];
  }
  __syncthreads();
  const int wave = tid >> 5, lane = tid & 31;
  const float scale = rsqrtf(1216.0f);
  for (int si = wave; si < 64; si += 8) {
    const int s = blockIdx.y*64 + si;
    const float* wrow = fc_w + ((size_t)lp*512 + s)*1216;
    float s0 = 0.0f, s1 = 0.0f;
    for (int k = lane; k < 1216; k += 32) {
      const float w = wrow[k];
      s0 += w*f0[k]; s1 += w*f1[k];
    }
#pragma unroll
    for (int off = 16; off > 0; off >>= 1) {
      s0 += __shfl_down(s0, off, 32);
      s1 += __shfl_down(s1, off, 32);
    }
    if (lane == 0) {
      const float bb = fc_b[lp*512 + s];
      out[(0*NLAY + l)*512 + s] = s0*scale + bb;
      out[(1*NLAY + l)*512 + s] = s1*scale + bb;
    }
  }
}

extern "C" void kernel_launch(void* const* d_in, const int* in_sizes, int n_in,
                              void* d_out, int out_size, void* d_ws, size_t ws_size,
                              hipStream_t stream)
{
  const int*   x       = (const int*)  d_in[0];
  const float* conv_w0 = (const float*)d_in[1];
  const float* conv_w  = (const float*)d_in[2];
  const float* conv_b  = (const float*)d_in[3];
  const float* prelu_a = (const float*)d_in[4];
  const float* fc_w    = (const float*)d_in[5];
  const float* fc_b    = (const float*)d_in[6];

  float* partials = (float*)d_ws;                               // NBLK * 43776 f32 (~22.4 MB)
  float* pooled   = partials + (size_t)NBLK*ACC_ELEMS;          // 2*18*19*64 f32
  float* zsnap    = pooled + (size_t)2*NLAY*19*64;              // 18*64 f32

  zsnap_kernel<<<1, 64, 0, stream>>>(conv_w, conv_b, prelu_a, zsnap);
  mlp_pool_kernel<<<NBLK, 256, LDS_BYTES, stream>>>(x, conv_w0, conv_w, conv_b, prelu_a, partials);
  reduce_pool_kernel<<<ACC_ELEMS/256, 256, 0, stream>>>(partials, zsnap, pooled);
  dim3 g(NLAY, 8);
  fc_kernel<<<g, 256, 0, stream>>>(pooled, fc_w, fc_b, (float*)d_out);

  (void)in_sizes; (void)n_in; (void)out_size; (void)ws_size;
}
